// CNN_medium_80092550135882
// MI455X (gfx1250) — 8-bit matrix operand order repaired and checked by reading, compiled, not run on hardware
//
#include <hip/hip_runtime.h>
#include <math.h>

typedef __attribute__((ext_vector_type(2))) int v2i;
typedef __attribute__((ext_vector_type(4))) int v4i;
typedef __attribute__((ext_vector_type(8))) int v8i;

#define BN_EPS 1e-5f

static __device__ __forceinline__ signed char sign8(float r) {
    return (r > 0.f) ? (signed char)1 : ((r < 0.f) ? (signed char)-1 : (signed char)0);
}

__global__ void k_zero(v4i* __restrict__ p, long n16) {
    long i = (long)blockIdx.x * blockDim.x + threadIdx.x;
    if (i < n16) { v4i z = {}; p[i] = z; }
}

__global__ void k_layer1(const float* __restrict__ x, const float* __restrict__ w1,
                         const float* __restrict__ b1,
                         const float* __restrict__ g, const float* __restrict__ b,
                         const float* __restrict__ m, const float* __restrict__ v,
                         signed char* __restrict__ out) {
    int idx = blockIdx.x * blockDim.x + threadIdx.x;
    int co = idx & 127;
    int p  = (idx >> 7) & 1023;
    int n  = idx >> 17;
    int y = p >> 5, xx = p & 31;
    float acc = 0.f;
    for (int ci = 0; ci < 3; ++ci) {
        const float* xp = x + (((long)(n * 3 + ci)) << 10);
        const float* wp = w1 + (co * 3 + ci) * 9;
        for (int ky = 0; ky < 3; ++ky) {
            int sy = y + ky - 1;
            if (sy < 0 || sy > 31) continue;
            for (int kx = 0; kx < 3; ++kx) {
                int sx = xx + kx - 1;
                if (sx < 0 || sx > 31) continue;
                float w = wp[ky * 3 + kx];
                float ws = (w > 0.f) ? 1.f : ((w < 0.f) ? -1.f : 0.f);
                acc += xp[(sy << 5) + sx] * ws;
            }
        }
    }
    float yv = acc + b1[co];
    float s = g[co] * rsqrtf(v[co] + BN_EPS);
    float r = (yv - m[co]) * s + b[co];
    out[(((long)n * 34 + (y + 1)) * 34 + (xx + 1)) * 128 + co] = sign8(r);
}

__global__ void k_packw(const float* __restrict__ w, signed char* __restrict__ wp,
                        int Cin, int total) {
    int idx = blockIdx.x * blockDim.x + threadIdx.x;
    if (idx >= total) return;
    int ci = idx % Cin;
    int kp = (idx / Cin) % 9;
    int co = idx / (Cin * 9);
    float wv = w[((long)(co * Cin + ci)) * 9 + kp];
    wp[idx] = sign8(wv);
}

__global__ void k_packw7(const float* __restrict__ w, signed char* __restrict__ wp) {
    int idx = blockIdx.x * blockDim.x + threadIdx.x;
    int ci = idx & 511;
    int kp = (idx >> 9) & 15;
    int co = idx >> 13;
    signed char r = 0;
    if (co < 10) r = sign8(w[((long)(co * 512 + ci)) * 16 + kp]);
    wp[idx] = r;
}

template<int CIN, int COUT, int H, int W>
__global__ void k_bconv(const signed char* __restrict__ in,
                        const signed char* __restrict__ wp,
                        int* __restrict__ out) {
    constexpr int PW = W + 2;
    constexpr int coPairs  = COUT >> 5;
    constexpr int pixTiles = (H * W) >> 4;
    int wave = (blockIdx.x * blockDim.x + threadIdx.x) >> 5;
    int lane = threadIdx.x & 31;
    int cp    = wave % coPairs;
    int t     = wave / coPairs;
    int pix_t = t % pixTiles;
    int n     = t / pixTiles;

    int mrow = lane & 15;
    int hi   = (lane >= 16) ? 1 : 0;
    int pix  = (pix_t << 4) + mrow;
    int py = pix / W, px = pix % W;
    int co0  = (cp << 5) + mrow;
    int co1  = co0 + 16;

    const signed char* imgBase =
        in + (((long)n * (H + 2) + py) * PW + px) * CIN + (hi ? 16 : 0);
    const signed char* b0Base = wp + ((long)co0 * 9) * CIN + (hi ? 16 : 0);
    const signed char* b1Base = wp + ((long)co1 * 9) * CIN + (hi ? 16 : 0);

    v8i acc0 = {}, acc1 = {};
    for (int ky = 0; ky < 3; ++ky) {
        for (int kx = 0; kx < 3; ++kx) {
            const signed char* ap  = imgBase + ((long)ky * PW + kx) * CIN;
            const signed char* bp0 = b0Base + (long)(ky * 3 + kx) * CIN;
            const signed char* bp1 = b1Base + (long)(ky * 3 + kx) * CIN;
#pragma unroll
            for (int cb = 0; cb < (CIN >> 6); ++cb) {
                const int ao = cb * 64;
                v2i a0 = *(const v2i*)(ap + ao);
                v2i a1 = *(const v2i*)(ap + ao + 8);
                v2i a2 = *(const v2i*)(ap + ao + 32);
                v2i a3 = *(const v2i*)(ap + ao + 40);
                v8i A; A[0]=a0.x; A[1]=a0.y; A[2]=a1.x; A[3]=a1.y;
                       A[4]=a2.x; A[5]=a2.y; A[6]=a3.x; A[7]=a3.y;
                v4i c0 = *(const v4i*)(bp0 + ao);
                v4i c1 = *(const v4i*)(bp0 + ao + 32);
                v8i B0; B0[0]=c0.x; B0[1]=c0.y; B0[2]=c0.z; B0[3]=c0.w;
                        B0[4]=c1.x; B0[5]=c1.y; B0[6]=c1.z; B0[7]=c1.w;
                v4i d0 = *(const v4i*)(bp1 + ao);
                v4i d1 = *(const v4i*)(bp1 + ao + 32);
                v8i B1; B1[0]=d0.x; B1[1]=d0.y; B1[2]=d0.z; B1[3]=d0.w;
                        B1[4]=d1.x; B1[5]=d1.y; B1[6]=d1.z; B1[7]=d1.w;
                acc0 = __builtin_amdgcn_wmma_i32_16x16x64_iu8(true, A, true, B0, acc0, false, false);
                acc1 = __builtin_amdgcn_wmma_i32_16x16x64_iu8(true, A, true, B1, acc1, false, false);
            }
        }
    }
    long obase = (long)n * (H * W) * COUT;
#pragma unroll
    for (int r = 0; r < 8; ++r) {
        long prow = (pix_t << 4) + r + (hi ? 8 : 0);
        out[obase + prow * COUT + co0] = acc0[r];
        out[obase + prow * COUT + co1] = acc1[r];
    }
}

__global__ void k_epilogue(const int* __restrict__ cin, const float* __restrict__ bias,
                           const float* __restrict__ g, const float* __restrict__ b,
                           const float* __restrict__ m, const float* __restrict__ v,
                           signed char* __restrict__ out,
                           int C, int Ho, int Wo, int pool, int padOut) {
    long idx = (long)blockIdx.x * blockDim.x + threadIdx.x;
    long total = (long)256 * Ho * Wo * C;
    if (idx >= total) return;
    int c = (int)(idx % C);
    long t = idx / C;
    int xo = (int)(t % Wo); t /= Wo;
    int yo = (int)(t % Ho);
    int n  = (int)(t / Ho);
    float yv;
    if (pool) {
        int Wc = Wo * 2;
        long base = (((long)n * (Ho * 2) + yo * 2) * Wc + xo * 2);
        int a0 = cin[(base) * C + c];
        int a1 = cin[(base + 1) * C + c];
        int a2 = cin[(base + Wc) * C + c];
        int a3 = cin[(base + Wc + 1) * C + c];
        int mx01 = a0 > a1 ? a0 : a1;
        int mx23 = a2 > a3 ? a2 : a3;
        int mx = mx01 > mx23 ? mx01 : mx23;
        yv = (float)mx + bias[c];
    } else {
        long base = ((long)n * Ho + yo) * Wo + xo;
        yv = (float)cin[base * C + c] + bias[c];
    }
    float s = g[c] * rsqrtf(v[c] + BN_EPS);
    float r = (yv - m[c]) * s + b[c];
    long oidx;
    if (padOut) {
        oidx = (((long)n * (Ho + 2) + (yo + 1)) * (Wo + 2) + (xo + 1)) * C + c;
    } else {
        oidx = idx;
    }
    out[oidx] = sign8(r);
}

__global__ void k_fc7(const signed char* __restrict__ a,
                      const signed char* __restrict__ wp,
                      int* __restrict__ out) {
    int wave = (blockIdx.x * blockDim.x + threadIdx.x) >> 5;
    int lane = threadIdx.x & 31;
    int mrow = lane & 15;
    int hi   = (lane >= 16) ? 1 : 0;
    const signed char* ap = a + ((long)(wave * 16 + mrow)) * 8192 + (hi ? 16 : 0);
    const signed char* bp = wp + (long)mrow * 8192 + (hi ? 16 : 0);
    v8i acc = {};
    for (int kk = 0; kk < 128; ++kk) {
        long o = (long)kk * 64;
        v2i a0 = *(const v2i*)(ap + o);
        v2i a1 = *(const v2i*)(ap + o + 8);
        v2i a2 = *(const v2i*)(ap + o + 32);
        v2i a3 = *(const v2i*)(ap + o + 40);
        v8i A; A[0]=a0.x; A[1]=a0.y; A[2]=a1.x; A[3]=a1.y;
               A[4]=a2.x; A[5]=a2.y; A[6]=a3.x; A[7]=a3.y;
        v4i b0 = *(const v4i*)(bp + o);
        v4i b1 = *(const v4i*)(bp + o + 32);
        v8i B; B[0]=b0.x; B[1]=b0.y; B[2]=b0.z; B[3]=b0.w;
               B[4]=b1.x; B[5]=b1.y; B[6]=b1.z; B[7]=b1.w;
        acc = __builtin_amdgcn_wmma_i32_16x16x64_iu8(true, A, true, B, acc, false, false);
    }
#pragma unroll
    for (int r = 0; r < 8; ++r) {
        int nrow = wave * 16 + r + (hi ? 8 : 0);
        out[nrow * 16 + mrow] = acc[r];
    }
}

__global__ void k_final(const int* __restrict__ sums, const float* __restrict__ b7,
                        const float* __restrict__ g, const float* __restrict__ b,
                        const float* __restrict__ m, const float* __restrict__ v,
                        float* __restrict__ out) {
    int n = blockIdx.x * blockDim.x + threadIdx.x;
    if (n >= 256) return;
    float z[10];
    float mx = -1e30f;
    for (int c = 0; c < 10; ++c) {
        float yv = (float)sums[n * 16 + c] + b7[c];
        float s = g[c] * rsqrtf(v[c] + BN_EPS);
        float r = (yv - m[c]) * s + b[c];
        z[c] = r;
        mx = fmaxf(mx, r);
    }
    float se = 0.f;
    for (int c = 0; c < 10; ++c) se += expf(z[c] - mx);
    float lse = logf(se);
    for (int c = 0; c < 10; ++c) out[n * 10 + c] = z[c] - mx - lse;
}

#define OFF_A1   0UL
#define OFF_A2   37879808UL
#define OFF_A3   48496640UL
#define OFF_A4   69730304UL
#define OFF_A5   76283904UL
#define OFF_A6   89391104UL
#define ZERO_BYTES 89391104UL
#define OFF_CONV 91488256UL
#define OFF_WP2  225705984UL
#define OFF_WP3  225853440UL
#define OFF_WP4  226148352UL
#define OFF_WP5  226738176UL
#define OFF_WP6  227917824UL
#define OFF_WP7  230277120UL
#define OFF_FC7  230408192UL

static inline void launch_epi(const int* cin, const float* bias, const float* const* bn,
                              signed char* out, int C, int Ho, int Wo, int pool, int padOut,
                              hipStream_t s) {
    long total = 256L * Ho * Wo * C;
    k_epilogue<<<(unsigned)((total + 255) / 256), 256, 0, s>>>(
        cin, bias, bn[0], bn[1], bn[2], bn[3], out, C, Ho, Wo, pool, padOut);
}

extern "C" void kernel_launch(void* const* d_in, const int* in_sizes, int n_in,
                              void* d_out, int out_size, void* d_ws, size_t ws_size,
                              hipStream_t stream) {
    (void)in_sizes; (void)n_in; (void)out_size; (void)ws_size;
    const float* x  = (const float*)d_in[0];
    const float* w[8]; const float* bconv[8];
    for (int j = 1; j <= 7; ++j) { w[j] = (const float*)d_in[2 * j - 1]; bconv[j] = (const float*)d_in[2 * j]; }
    const float* bn[7][4];
    for (int j = 1; j <= 6; ++j)
        for (int q = 0; q < 4; ++q) bn[j][q] = (const float*)d_in[15 + (j - 1) * 4 + q];
    const float* bnf[4] = { (const float*)d_in[39], (const float*)d_in[40],
                            (const float*)d_in[41], (const float*)d_in[42] };

    char* ws = (char*)d_ws;
    signed char* a1 = (signed char*)(ws + OFF_A1);
    signed char* a2 = (signed char*)(ws + OFF_A2);
    signed char* a3 = (signed char*)(ws + OFF_A3);
    signed char* a4 = (signed char*)(ws + OFF_A4);
    signed char* a5 = (signed char*)(ws + OFF_A5);
    signed char* a6 = (signed char*)(ws + OFF_A6);
    int*  conv32 = (int*)(ws + OFF_CONV);
    signed char* wp2 = (signed char*)(ws + OFF_WP2);
    signed char* wp3 = (signed char*)(ws + OFF_WP3);
    signed char* wp4 = (signed char*)(ws + OFF_WP4);
    signed char* wp5 = (signed char*)(ws + OFF_WP5);
    signed char* wp6 = (signed char*)(ws + OFF_WP6);
    signed char* wp7 = (signed char*)(ws + OFF_WP7);
    int* fc7 = (int*)(ws + OFF_FC7);

    {
        long n16 = ZERO_BYTES / 16;
        k_zero<<<(unsigned)((n16 + 255) / 256), 256, 0, stream>>>((v4i*)ws, n16);
    }

    k_packw<<<576,  256, 0, stream>>>(w[2], wp2, 128, 128 * 9 * 128);
    k_packw<<<1152, 256, 0, stream>>>(w[3], wp3, 128, 256 * 9 * 128);
    k_packw<<<2304, 256, 0, stream>>>(w[4], wp4, 256, 256 * 9 * 256);
    k_packw<<<4608, 256, 0, stream>>>(w[5], wp5, 256, 512 * 9 * 256);
    k_packw<<<9216, 256, 0, stream>>>(w[6], wp6, 512, 512 * 9 * 512);
    k_packw7<<<512, 256, 0, stream>>>(w[7], wp7);

    k_layer1<<<131072, 256, 0, stream>>>(x, w[1], bconv[1],
                                         bn[1][0], bn[1][1], bn[1][2], bn[1][3], a1);

    k_bconv<128,128,32,32><<<8192, 256, 0, stream>>>(a1, wp2, conv32);
    launch_epi(conv32, bconv[2], bn[2], a2, 128, 16, 16, 1, 1, stream);
    k_bconv<128,256,16,16><<<4096, 256, 0, stream>>>(a2, wp3, conv32);
    launch_epi(conv32, bconv[3], bn[3], a3, 256, 16, 16, 0, 1, stream);
    k_bconv<256,256,16,16><<<4096, 256, 0, stream>>>(a3, wp4, conv32);
    launch_epi(conv32, bconv[4], bn[4], a4, 256, 8, 8, 1, 1, stream);
    k_bconv<256,512,8,8><<<2048, 256, 0, stream>>>(a4, wp5, conv32);
    launch_epi(conv32, bconv[5], bn[5], a5, 512, 8, 8, 0, 1, stream);
    k_bconv<512,512,8,8><<<2048, 256, 0, stream>>>(a5, wp6, conv32);
    launch_epi(conv32, bconv[6], bn[6], a6, 512, 4, 4, 1, 0, stream);

    k_fc7<<<2, 256, 0, stream>>>(a6, wp7, fc7);
    k_final<<<1, 256, 0, stream>>>(fc7, bconv[7], bnf[0], bnf[1], bnf[2], bnf[3],
                                   (float*)d_out);
}
